// SSelfattention_22058952032500
// MI455X (gfx1250) — matrix operand order repaired and checked by reading, not compiled, not run on hardware
//
#include <hip/hip_runtime.h>

#define B_  4
#define N_  128
#define T_  256
#define C_  256
#define H_  8
#define D_  32

typedef __attribute__((ext_vector_type(16))) __bf16 v16bf;
typedef __attribute__((ext_vector_type(8)))  __bf16 v8bf;
typedef __attribute__((ext_vector_type(8)))  float  v8f;
typedef __attribute__((ext_vector_type(4)))  float  v4f;

// LDS strides in bf16 elements; padded so consecutive rows shift banks.
#define XS  264  // Ob: 128 x 264  (concat head outputs, row-major [node][c])
#define HS  40   // Xvh/Xkh/Qh/Kh: 128 x 40 (per-head 128x32, row-major [node][d])
#define VTS 136  // Vt: 32 x 136  (V TRANSPOSED: [d][node])
#define SS  136  // Sb: 128 x 136 (scores / attention, row-major [q][k])

#define OFF_OB   0        // 67584
#define OFF_XVH  67584    // +10240
#define OFF_XKH  77824    // +10240
#define OFF_QH   88064    // +10240
#define OFF_KH   98304    // +10240
#define OFF_VT   108544   // +8704
#define OFF_SB   117248   // +34816
#define OFF_RED  152064   // +1024
#define OFF_SUM  153088   // +1024
#define SMEM_BYTES 154112 // 150.5 KB -> 2 workgroups per 320 KB WGP

__device__ __forceinline__ v8f wmma_bf16(v16bf a, v16bf b, v8f c) {
  return __builtin_amdgcn_wmma_f32_16x16x32_bf16(
      /*neg_a=*/false, a, /*neg_b=*/false, b,
      /*c_mod=*/(short)0, c, /*reuse_a=*/false, /*reuse_b=*/false);
}

// A-matrix 16x32 bf16 fragment (ISA 7.12.2): m = lane&15,
// halves 0..7 -> K=koff+0..7, halves 8..15 -> K=16+koff+0..7, koff=(lane>=16)*8.
// Two contiguous 16B chunks -> 2x ds_load_b128.
__device__ __forceinline__ v16bf load_a_frag(const __bf16* X, int rs, int row0,
                                             int col0, int lane) {
  const __bf16* p = X + (row0 + (lane & 15)) * rs + col0 + ((lane & 16) >> 1);
  v16bf a;
#pragma unroll
  for (int i = 0; i < 8; ++i) a[i] = p[i];
#pragma unroll
  for (int i = 0; i < 8; ++i) a[8 + i] = p[16 + i];
  return a;
}

// A fragment from global f32 rows with on-the-fly bf16 conversion.
__device__ __forceinline__ v16bf load_a_frag_g(const float* X, int rs, int row0,
                                               int col0, int lane) {
  const float* p = X + (row0 + (lane & 15)) * rs + col0 + ((lane & 16) >> 1);
  v16bf a;
#pragma unroll
  for (int i = 0; i < 8; ++i) a[i] = (__bf16)p[i];
#pragma unroll
  for (int i = 0; i < 8; ++i) a[8 + i] = (__bf16)p[16 + i];
  return a;
}

// B-matrix 32x16 fragment of X^T: B[kk][nn] = X[n0+nn][k0+kk].
// Per-lane two 8-half chunks (K=koff+0..7, 16+koff+0..7) -> 2x ds_load_b128.
__device__ __forceinline__ v16bf load_bT_frag(const __bf16* X, int rs, int n0,
                                              int k0, int lane) {
  const __bf16* p = X + (n0 + (lane & 15)) * rs + k0 + ((lane & 16) >> 1);
  v16bf b;
#pragma unroll
  for (int i = 0; i < 8; ++i) b[i] = p[i];
#pragma unroll
  for (int i = 0; i < 8; ++i) b[8 + i] = p[16 + i];
  return b;
}

// Same, from global f32 with conversion (for Wv^T fragments).
__device__ __forceinline__ v16bf load_bT_frag_g(const float* X, int rs, int n0,
                                                int k0, int lane) {
  const float* p = X + (n0 + (lane & 15)) * rs + k0 + ((lane & 16) >> 1);
  v16bf b;
#pragma unroll
  for (int i = 0; i < 8; ++i) b[i] = (__bf16)p[i];
#pragma unroll
  for (int i = 0; i < 8; ++i) b[8 + i] = (__bf16)p[16 + i];
  return b;
}

// C/D f32 layout: VGPR r, lane L -> M = r + 8*(L>=16), N = L&15.
// TRANSPOSED store X[n][m]: per lane the 8 M-values are contiguous ->
// pack to 16B and emit a single ds_store_b128.
__device__ __forceinline__ void store_cd_bf16_T(__bf16* X, int rs, int row0,
                                                int col0, int lane, v8f c,
                                                float scale) {
  v8bf pk;
#pragma unroll
  for (int r = 0; r < 8; ++r) pk[r] = (__bf16)(c[r] * scale);
  __bf16* p = X + (col0 + (lane & 15)) * rs + row0 + ((lane & 16) >> 1);
  *(v8bf*)p = pk;
}

__global__ __launch_bounds__(256)
void attn_fused_kernel(const float* __restrict__ values,
                       const float* __restrict__ keys,
                       const float* __restrict__ Wq,
                       const float* __restrict__ Wk,
                       const float* __restrict__ Wv,
                       const float* __restrict__ Wfc,
                       const float* __restrict__ bfc,
                       float* __restrict__ out) {
  extern __shared__ char smem[];
  __bf16* Ob   = (__bf16*)(smem + OFF_OB);    // 128x256 concat head outputs
  __bf16* Xvh  = (__bf16*)(smem + OFF_XVH);   // per-head values slice 128x32
  __bf16* Xkh  = (__bf16*)(smem + OFF_XKH);   // per-head keys   slice 128x32
  __bf16* Qh   = (__bf16*)(smem + OFF_QH);    // Q 128x32 row-major [node][e]
  __bf16* Kh   = (__bf16*)(smem + OFF_KH);    // K 128x32 row-major [node][e]
  __bf16* Vt   = (__bf16*)(smem + OFF_VT);    // V TRANSPOSED 32x128 [e][node]
  __bf16* Sb   = (__bf16*)(smem + OFF_SB);    // scores/attn 128x128 [q][k]
  float*  red  = (float*)(smem + OFF_RED);    // 2x128 partial max
  float*  sums = (float*)(smem + OFF_SUM);    // 2x128 partial sum

  const int tid  = threadIdx.x;
  const int lane = tid & 31;
  const int wave = tid >> 5;
  const int b = blockIdx.x >> 8;   // blockIdx = b*T + t
  const int t = blockIdx.x & 255;
  const size_t base = ((size_t)(b * N_) * T_ + t) * C_;

  // Wq/Wk live as A-fragments (Q^T = Wq X^T), Wv as B-fragments (V = X Wv^T).
  v16bf wqA[2], wkA[2], wvB[2];
#pragma unroll
  for (int et = 0; et < 2; ++et) {
    wqA[et] = load_a_frag_g(Wq, D_, et * 16, 0, lane);
    wkA[et] = load_a_frag_g(Wk, D_, et * 16, 0, lane);
    wvB[et] = load_bT_frag_g(Wv, D_, et * 16, 0, lane);
  }

  for (int h = 0; h < H_; ++h) {
    // ---- Stage 0: stage THIS head's 128x32 slices of values/keys ----
    // Each element of values/keys is read from HBM exactly once overall.
#pragma unroll 4
    for (int i = 0; i < 16; ++i) {
      const int r = wave * 16 + i;
      const size_t g = base + (size_t)r * (T_ * C_) + h * D_ + lane;
      Xvh[r * HS + lane] = (__bf16)values[g];
      Xkh[r * HS + lane] = (__bf16)keys[g];
    }
    // Prefetch next head's slices into cache while we compute this head.
    if (h + 1 < H_) {
      const int r = tid & 127;
      const float* pf = ((tid < 128) ? values : keys) +
                        base + (size_t)r * (T_ * C_) + (h + 1) * D_;
      __builtin_prefetch(pf, 0, 1);
    }
    __syncthreads();

    // ---- Stage 1: projections. Wave owns node-band `wave` (16 rows) ----
    // Q^T = Wq Xv^T and K^T = Wk Xk^T: transposed C/D store lands row-major
    // Q[node][e] / K[node][e] as a single ds_store_b128 per tile.
    {
      v16bf bxv = load_bT_frag(Xvh, HS, wave * 16, 0, lane);  // Xv^T frag
      v16bf bxk = load_bT_frag(Xkh, HS, wave * 16, 0, lane);  // Xk^T frag
#pragma unroll
      for (int et = 0; et < 2; ++et) {
        v8f cq = {};
        cq = wmma_bf16(wqA[et], bxv, cq);
        store_cd_bf16_T(Qh, HS, et * 16, wave * 16, lane, cq, 1.0f);
        v8f ck = {};
        ck = wmma_bf16(wkA[et], bxk, ck);
        store_cd_bf16_T(Kh, HS, et * 16, wave * 16, lane, ck, 1.0f);
      }
      // V = Xv Wv^T, stored transposed -> Vt[e][node] (A-layout for stage 4).
      v16bf axv = load_a_frag(Xvh, HS, wave * 16, 0, lane);
#pragma unroll
      for (int et = 0; et < 2; ++et) {
        v8f cv = {};
        cv = wmma_bf16(axv, wvB[et], cv);
        store_cd_bf16_T(Vt, VTS, wave * 16, et * 16, lane, cv, 1.0f);
      }
    }
    __syncthreads();

    // ---- Stage 2: S^T = K Q^T; transposed store -> row-major S[q][k] ----
    {
      v16bf ak = load_a_frag(Kh, HS, wave * 16, 0, lane);     // K band (k rows)
#pragma unroll
      for (int qt = 0; qt < 8; ++qt) {
        v16bf bq = load_bT_frag(Qh, HS, qt * 16, 0, lane);    // Q^T fragment
        v8f c = {};
        c = wmma_bf16(ak, bq, c);
        // D = S^T tile (m=k, n=q); T-store writes S[q][k], scaled 1/sqrt(256).
        store_cd_bf16_T(Sb, SS, wave * 16, qt * 16, lane, c, 0.0625f);
      }
    }
    __syncthreads();

    // ---- Stage 3: softmax over the QUERY axis (per column k of S[q][k]) ----
    {
      const int col = tid & 127;
      const int r0  = (tid >> 7) * 64;       // two threads split each column
      float m = -3.0e38f;
      for (int r = 0; r < 64; ++r)
        m = fmaxf(m, (float)Sb[(r0 + r) * SS + col]);
      red[(tid >> 7) * 128 + col] = m;
      __syncthreads();
      const float mm = fmaxf(red[col], red[128 + col]);
      float s = 0.0f;
      for (int r = 0; r < 64; ++r) {
        const float e = __expf((float)Sb[(r0 + r) * SS + col] - mm);
        Sb[(r0 + r) * SS + col] = (__bf16)e;   // in-place, disjoint per thread
        s += e;
      }
      sums[(tid >> 7) * 128 + col] = s;
      __syncthreads();
      const float inv = 1.0f / (sums[col] + sums[128 + col]);
      for (int r = 0; r < 64; ++r)
        Sb[(r0 + r) * SS + col] = (__bf16)((float)Sb[(r0 + r) * SS + col] * inv);
    }
    __syncthreads();

    // ---- Stage 4: O^T = V^T A^T; transposed store -> row-major Ob[q][c] ----
    {
      v16bf bs[4];                            // A^T fragments, wave's q-band
#pragma unroll
      for (int ks = 0; ks < 4; ++ks)
        bs[ks] = load_bT_frag(Sb, SS, wave * 16, ks * 32, lane);
#pragma unroll
      for (int et = 0; et < 2; ++et) {
        v8f c = {};
#pragma unroll
        for (int ks = 0; ks < 4; ++ks) {
          v16bf av = load_a_frag(Vt, VTS, et * 16, ks * 32, lane);  // V^T
          c = wmma_bf16(av, bs[ks], c);
        }
        store_cd_bf16_T(Ob, XS, h * D_ + et * 16, wave * 16, lane, c, 1.0f);
      }
    }
    __syncthreads();
  }

  // ---- Stage 5: out^T = Wfc Ob^T + bfc. Wave owns node-band `wave` ----
  // B-fragments (Ob^T) hoisted once; Wfc rows stream as A-frags from L2.
  // Each lane ends with 8 CONSECUTIVE channels of one row -> b128 stores.
  v16bf bo[8];
#pragma unroll
  for (int ks = 0; ks < 8; ++ks)
    bo[ks] = load_bT_frag(Ob, XS, wave * 16, ks * 32, lane);
  const int nrow = wave * 16 + (lane & 15);       // output node row
  float* orow = out + base + (size_t)nrow * (T_ * C_);
  for (int mt = 0; mt < 16; ++mt) {               // 16 channel tiles
    v8f c = {};
#pragma unroll
    for (int ks = 0; ks < 8; ++ks) {
      v16bf aw = load_a_frag_g(Wfc, C_, mt * 16, ks * 32, lane);
      c = wmma_bf16(aw, bo[ks], c);
    }
    const int cc0 = mt * 16 + ((lane & 16) >> 1); // 8 consecutive channels
    const v4f* pb = (const v4f*)(bfc + cc0);
    v4f lo = pb[0], hi = pb[1];
    v4f o0 = {c[0] + lo[0], c[1] + lo[1], c[2] + lo[2], c[3] + lo[3]};
    v4f o1 = {c[4] + hi[0], c[5] + hi[1], c[6] + hi[2], c[7] + hi[3]};
    *(v4f*)(orow + cc0)     = o0;
    *(v4f*)(orow + cc0 + 4) = o1;
  }
}

extern "C" void kernel_launch(void* const* d_in, const int* in_sizes, int n_in,
                              void* d_out, int out_size, void* d_ws, size_t ws_size,
                              hipStream_t stream) {
  (void)in_sizes; (void)n_in; (void)out_size; (void)d_ws; (void)ws_size;
  const float* values = (const float*)d_in[0];
  const float* keys   = (const float*)d_in[1];
  // d_in[2] = query: shape-only in the reference (original code bug) -> unused.
  const float* Wq  = (const float*)d_in[3];
  const float* Wk  = (const float*)d_in[4];
  const float* Wv  = (const float*)d_in[5];
  const float* Wfc = (const float*)d_in[6];
  const float* bfc = (const float*)d_in[7];
  float* out = (float*)d_out;

  hipFuncSetAttribute(reinterpret_cast<const void*>(attn_fused_kernel),
                      hipFuncAttributeMaxDynamicSharedMemorySize, SMEM_BYTES);
  attn_fused_kernel<<<dim3(B_ * T_), dim3(256), SMEM_BYTES, stream>>>(
      values, keys, Wq, Wk, Wv, Wfc, bfc, out);
}
